// SelfAttention_81466939671203
// MI455X (gfx1250) — matrix operand order repaired and checked by reading, not compiled, not run on hardware
//
#include <hip/hip_runtime.h>
#include <hip/hip_bf16.h>

// ---------------------------------------------------------------------------
// Self-attention (B=2,S=2048,D=2048,H=16,DH=128) for gfx1250.
// All GEMMs via v_wmma_f32_16x16x32_bf16 (fp32 accumulate).
// Pipeline: cvt(x) + transpose-cvt(Wqkv,Wo) -> WMMA GEMM (qkv) -> RoPE/layout
//           -> flash-attention (WMMA scores + WMMA P@V, online softmax)
//           -> WMMA GEMM (out proj, fp32 out).
// ---------------------------------------------------------------------------

typedef __attribute__((ext_vector_type(16))) __bf16 v16bf;
typedef __attribute__((ext_vector_type(8)))  __bf16 bf16x8;
typedef __attribute__((ext_vector_type(8)))  float  v8f;

constexpr int Bz  = 2;
constexpr int Sd  = 2048;
constexpr int Dd  = 2048;
constexpr int Hh  = 16;
constexpr int DHd = 128;
constexpr int Mrows = Bz * Sd;   // 4096
constexpr int N3    = 3 * Dd;    // 6144

__device__ __forceinline__ v16bf combine8(bf16x8 lo, bf16x8 hi) {
  v16bf r;
#pragma unroll
  for (int i = 0; i < 8; ++i) { r[i] = lo[i]; r[i + 8] = hi[i]; }
  return r;
}
// A-fragment (16x32 bf16): lane<16 -> row M=lane&15, K {k..k+7, k+16..k+23};
// lane>=16 -> K {k+8..k+15, k+24..k+31}. Caller passes p = &A[row][k + lhalf*8].
__device__ __forceinline__ v16bf load_a_frag(const __bf16* p) {
  bf16x8 lo = *(const bf16x8*)(p);
  bf16x8 hi = *(const bf16x8*)(p + 16);
  return combine8(lo, hi);
}
// B-fragment (32x16 bf16), Bt stored [N][K]: lane -> col N=lane&15,
// K {k+lhalf*8..+7, k+16+lhalf*8..+7}. Caller passes p = &Bt[n][k + lhalf*8].
__device__ __forceinline__ v16bf load_b_frag(const __bf16* p) {
  bf16x8 lo = *(const bf16x8*)(p);
  bf16x8 hi = *(const bf16x8*)(p + 16);
  return combine8(lo, hi);
}
__device__ __forceinline__ v8f wmma_bf16(v16bf a, v16bf b, v8f c) {
  return __builtin_amdgcn_wmma_f32_16x16x32_bf16(
      /*neg_a=*/false, a, /*neg_b=*/false, b,
      /*c_mod=*/(short)0, c, /*reuse_a=*/false, /*reuse_b=*/false);
}
__device__ __forceinline__ v8f v8f_zero() {
  v8f z = {0.f, 0.f, 0.f, 0.f, 0.f, 0.f, 0.f, 0.f};
  return z;
}

// --------------------------- fp32 -> bf16 copy ------------------------------
__global__ void cvt_f32_bf16_kernel(const float* __restrict__ in,
                                    __bf16* __restrict__ out, int n) {
  int i = blockIdx.x * blockDim.x + threadIdx.x;
  if (i < n) out[i] = (__bf16)in[i];
}

// ---------------- transpose + convert: in[R][C] f32 -> out[C][R] bf16 -------
__global__ __launch_bounds__(256)
void transpose_f32_bf16_kernel(const float* __restrict__ in,
                               __bf16* __restrict__ out, int R, int C) {
  __shared__ float tile[32][33];
  int tx = threadIdx.x, ty = threadIdx.y;           // (32, 8)
  int c0 = blockIdx.x * 32, r0 = blockIdx.y * 32;
#pragma unroll
  for (int i = 0; i < 4; ++i)
    tile[ty + 8 * i][tx] = in[(size_t)(r0 + ty + 8 * i) * C + (c0 + tx)];
  __syncthreads();
#pragma unroll
  for (int i = 0; i < 4; ++i)
    out[(size_t)(c0 + ty + 8 * i) * R + (r0 + tx)] = (__bf16)tile[tx][ty + 8 * i];
}

// ---------------- WMMA GEMM: C[M][N] = A[M][K] * Bt[N][K]^T + bias[N] -------
// 128 threads = 4 waves; block tile 128x128; wave tile 64x64 (4x4 WMMA tiles).
__global__ __launch_bounds__(128)
void gemm_bf16_nt_kernel(const __bf16* __restrict__ A,
                         const __bf16* __restrict__ Bt,
                         const float* __restrict__ bias,
                         float* __restrict__ C, int Mm, int Nn, int Kk) {
  int wave = threadIdx.x >> 5;
  int lane = threadIdx.x & 31;
  int lrow = lane & 15;
  int lhalf = lane >> 4;
  int wm = blockIdx.y * 128 + (wave & 1) * 64;
  int wn = blockIdx.x * 128 + (wave >> 1) * 64;

  const __bf16* Ap = A  + (size_t)(wm + lrow) * Kk + lhalf * 8;
  const __bf16* Bp = Bt + (size_t)(wn + lrow) * Kk + lhalf * 8;

  v8f acc[4][4];
#pragma unroll
  for (int mi = 0; mi < 4; ++mi)
#pragma unroll
    for (int ni = 0; ni < 4; ++ni) acc[mi][ni] = v8f_zero();

  for (int k0 = 0; k0 < Kk; k0 += 32) {
    v16bf a[4], b[4];
#pragma unroll
    for (int mi = 0; mi < 4; ++mi)
      a[mi] = load_a_frag(Ap + (size_t)mi * 16 * Kk + k0);
#pragma unroll
    for (int ni = 0; ni < 4; ++ni)
      b[ni] = load_b_frag(Bp + (size_t)ni * 16 * Kk + k0);
#pragma unroll
    for (int mi = 0; mi < 4; ++mi)
#pragma unroll
      for (int ni = 0; ni < 4; ++ni)
        acc[mi][ni] = wmma_bf16(a[mi], b[ni], acc[mi][ni]);
  }

#pragma unroll
  for (int mi = 0; mi < 4; ++mi)
#pragma unroll
    for (int ni = 0; ni < 4; ++ni) {
      int c = wn + ni * 16 + lrow;
      float bv = bias[c];
#pragma unroll
      for (int v = 0; v < 8; ++v) {
        int r = wm + mi * 16 + v + 8 * lhalf;
        C[(size_t)r * Nn + c] = acc[mi][ni][v] + bv;
      }
    }
}

// ---------------- RoPE + head layout --------------------------------------
// qkv f32 [B,S,3D] -> q,k bf16 [B,H,S,DH] (roped), v bf16 [B,H,DH,S] (transposed)
__global__ void rope_split_kernel(const float* __restrict__ qkv,
                                  __bf16* __restrict__ qb,
                                  __bf16* __restrict__ kb,
                                  __bf16* __restrict__ vt) {
  int tid = blockIdx.x * blockDim.x + threadIdx.x;  // B*S*H*64 threads
  int j = tid & 63;
  int h = (tid >> 6) & (Hh - 1);
  int s = (tid >> 10) & (Sd - 1);
  int b = tid >> 21;

  const float* base = qkv + ((size_t)(b * Sd + s)) * N3 + h * DHd;
  // freq_j = s * 10000^{-2j/DH}
  float f = (float)s * __expf(-(2.0f * (float)j / (float)DHd) * 9.210340371976184f);
  float cs = __cosf(f), sn = __sinf(f);

  size_t qo = ((size_t)(b * Hh + h) * Sd + s) * DHd;
  // q
  {
    float u0 = base[j], u1 = base[64 + j];
    float ue = base[2 * j], uo = base[2 * j + 1];
    qb[qo + j]      = (__bf16)(u0 * cs + uo * sn);
    qb[qo + 64 + j] = (__bf16)(u1 * cs - ue * sn);
  }
  // k
  {
    const float* kbase = base + Dd;
    float u0 = kbase[j], u1 = kbase[64 + j];
    float ue = kbase[2 * j], uo = kbase[2 * j + 1];
    kb[qo + j]      = (__bf16)(u0 * cs + uo * sn);
    kb[qo + 64 + j] = (__bf16)(u1 * cs - ue * sn);
  }
  // v transposed: vt[(b*H+h)][d][s]
  {
    const float* vbase = base + 2 * Dd;
    size_t vo = (size_t)(b * Hh + h) * DHd * Sd;
    vt[vo + (size_t)j * Sd + s]        = (__bf16)vbase[j];
    vt[vo + (size_t)(64 + j) * Sd + s] = (__bf16)vbase[64 + j];
  }
}

// ---------------- flash attention ------------------------------------------
// grid (S/64, B*H); 128 threads = 4 waves; each wave owns 16 query rows.
__global__ __launch_bounds__(128)
void attn_kernel(const __bf16* __restrict__ Q, const __bf16* __restrict__ K,
                 const __bf16* __restrict__ Vt, __bf16* __restrict__ Oo) {
  __shared__ __bf16 Pl[4][16][32];   // per-wave P tile (C-layout -> A-layout relay)

  int bh = blockIdx.y;
  int wave = threadIdx.x >> 5;
  int lane = threadIdx.x & 31;
  int lrow = lane & 15;
  int lhalf = lane >> 4;
  int qbase = blockIdx.x * 64 + wave * 16;

  const __bf16* Qb = Q  + (size_t)bh * Sd * DHd;
  const __bf16* Kb = K  + (size_t)bh * Sd * DHd;
  const __bf16* Vb = Vt + (size_t)bh * DHd * Sd;

  // Q fragments (row qbase+lrow), held in registers for the whole k-loop.
  v16bf qf[4];
  const __bf16* qp = Qb + (size_t)(qbase + lrow) * DHd + lhalf * 8;
#pragma unroll
  for (int kc = 0; kc < 4; ++kc) qf[kc] = load_a_frag(qp + kc * 32);

  v8f o[8];
#pragma unroll
  for (int t = 0; t < 8; ++t) o[t] = v8f_zero();
  float m[8], l[8];
#pragma unroll
  for (int v = 0; v < 8; ++v) { m[v] = -1e30f; l[v] = 0.f; }

  const float scale = 0.08838834764831845f;  // 1/sqrt(128)
  __bf16* pw = &Pl[wave][0][0];

  for (int kbk = 0; kbk < Sd; kbk += 32) {
    // ---- scores: two 16x16 tiles over this 32-key block ----
    v8f s0 = v8f_zero(), s1 = v8f_zero();
    const __bf16* kp = Kb + (size_t)(kbk + lrow) * DHd + lhalf * 8;
#pragma unroll
    for (int kc = 0; kc < 4; ++kc) {
      v16bf b0 = load_b_frag(kp + kc * 32);
      v16bf b1 = load_b_frag(kp + (size_t)16 * DHd + kc * 32);
      s0 = wmma_bf16(qf[kc], b0, s0);
      s1 = wmma_bf16(qf[kc], b1, s1);
    }

    // ---- online softmax (rows live as: VGPR v -> row v+8*lhalf; 16 lanes/row)
    float p0[8], p1[8], mn[8], rs[8];
#pragma unroll
    for (int v = 0; v < 8; ++v) {
      p0[v] = s0[v] * scale;
      p1[v] = s1[v] * scale;
      mn[v] = fmaxf(p0[v], p1[v]);
    }
#pragma unroll
    for (int mask = 1; mask <= 8; mask <<= 1)
#pragma unroll
      for (int v = 0; v < 8; ++v)
        mn[v] = fmaxf(mn[v], __shfl_xor(mn[v], mask, 32));
#pragma unroll
    for (int v = 0; v < 8; ++v) {
      mn[v] = fmaxf(mn[v], m[v]);
      float alpha = __expf(m[v] - mn[v]);
      m[v] = mn[v];
      p0[v] = __expf(p0[v] - mn[v]);
      p1[v] = __expf(p1[v] - mn[v]);
      rs[v] = p0[v] + p1[v];
      l[v] *= alpha;
#pragma unroll
      for (int t = 0; t < 8; ++t) o[t][v] *= alpha;
    }
#pragma unroll
    for (int mask = 1; mask <= 8; mask <<= 1)
#pragma unroll
      for (int v = 0; v < 8; ++v)
        rs[v] += __shfl_xor(rs[v], mask, 32);
#pragma unroll
    for (int v = 0; v < 8; ++v) l[v] += rs[v];

    // ---- C-layout P -> LDS -> A-fragment layout ----
#pragma unroll
    for (int v = 0; v < 8; ++v) {
      pw[(v + 8 * lhalf) * 32 + lrow]      = (__bf16)p0[v];
      pw[(v + 8 * lhalf) * 32 + 16 + lrow] = (__bf16)p1[v];
    }
    asm volatile("s_wait_dscnt 0" ::: "memory");
    v16bf pf = load_a_frag(pw + lrow * 32 + lhalf * 8);

    // ---- O += P @ V  (V as B-operand from [DH][S] layout) ----
    const __bf16* vp = Vb + (size_t)lrow * Sd + kbk + lhalf * 8;
#pragma unroll
    for (int ni = 0; ni < 8; ++ni) {
      v16bf bv = load_b_frag(vp + (size_t)ni * 16 * Sd);
      o[ni] = wmma_bf16(pf, bv, o[ni]);
    }
  }

  // ---- epilogue: normalize, write [B,S,D] bf16 (col = h*DH + d) ----
  int b = bh >> 4, h = bh & 15;
#pragma unroll
  for (int v = 0; v < 8; ++v) {
    float inv = 1.0f / l[v];
    int q = qbase + v + 8 * lhalf;
    size_t rowoff = ((size_t)(b * Sd + q)) * Dd + h * DHd;
#pragma unroll
    for (int ni = 0; ni < 8; ++ni)
      Oo[rowoff + ni * 16 + lrow] = (__bf16)(o[ni][v] * inv);
  }
}

// ---------------------------------------------------------------------------
extern "C" void kernel_launch(void* const* d_in, const int* in_sizes, int n_in,
                              void* d_out, int out_size, void* d_ws, size_t ws_size,
                              hipStream_t stream) {
  const float* x    = (const float*)d_in[0];
  const float* Wqkv = (const float*)d_in[1];
  const float* bqkv = (const float*)d_in[2];
  const float* Wo   = (const float*)d_in[3];
  const float* bo   = (const float*)d_in[4];
  float* out = (float*)d_out;

  char* w = (char*)d_ws;
  size_t off = 0;
  __bf16* x_bf   = (__bf16*)(w + off); off += (size_t)Mrows * Dd * 2;
  __bf16* wqkv_t = (__bf16*)(w + off); off += (size_t)N3 * Dd * 2;
  __bf16* wo_t   = (__bf16*)(w + off); off += (size_t)Dd * Dd * 2;
  float*  qkv    = (float*) (w + off); off += (size_t)Mrows * N3 * 4;
  __bf16* q_bf   = (__bf16*)(w + off); off += (size_t)Bz * Hh * Sd * DHd * 2;
  __bf16* k_bf   = (__bf16*)(w + off); off += (size_t)Bz * Hh * Sd * DHd * 2;
  __bf16* vt_bf  = (__bf16*)(w + off); off += (size_t)Bz * Hh * DHd * Sd * 2;
  __bf16* ao_bf  = (__bf16*)(w + off); off += (size_t)Mrows * Dd * 2;
  (void)ws_size; (void)n_in; (void)in_sizes; (void)out_size;

  // 1) precision/layout conversion
  {
    int n = Mrows * Dd;
    cvt_f32_bf16_kernel<<<(n + 255) / 256, 256, 0, stream>>>(x, x_bf, n);
    transpose_f32_bf16_kernel<<<dim3(N3 / 32, Dd / 32), dim3(32, 8), 0, stream>>>(
        Wqkv, wqkv_t, Dd, N3);
    transpose_f32_bf16_kernel<<<dim3(Dd / 32, Dd / 32), dim3(32, 8), 0, stream>>>(
        Wo, wo_t, Dd, Dd);
  }
  // 2) qkv = x @ Wqkv + bqkv  (fp32 out)
  gemm_bf16_nt_kernel<<<dim3(N3 / 128, Mrows / 128), 128, 0, stream>>>(
      x_bf, wqkv_t, bqkv, qkv, Mrows, N3, Dd);
  // 3) RoPE + per-head layout (q,k roped bf16; v transposed bf16)
  {
    int n = Bz * Sd * Hh * 64;
    rope_split_kernel<<<n / 256, 256, 0, stream>>>(qkv, q_bf, k_bf, vt_bf);
  }
  // 4) flash attention
  attn_kernel<<<dim3(Sd / 64, Bz * Hh), 128, 0, stream>>>(q_bf, k_bf, vt_bf, ao_bf);
  // 5) out = attn @ Wo + bo (fp32 out)
  gemm_bf16_nt_kernel<<<dim3(Dd / 128, Mrows / 128), 128, 0, stream>>>(
      ao_bf, wo_t, bo, out, Mrows, Dd, Dd);
}
